// MultiHeadAttention_74448963109519
// MI455X (gfx1250) — matrix operand order repaired and checked by reading, not compiled, not run on hardware
//
#include <hip/hip_runtime.h>
#include <hip/hip_bf16.h>
#include <math.h>

typedef __attribute__((ext_vector_type(16))) __bf16 v16bf;
typedef __attribute__((ext_vector_type(8)))  float  v8f;
typedef __attribute__((ext_vector_type(4)))  unsigned int v4u;
typedef __attribute__((ext_vector_type(4)))  unsigned int su4;
typedef __attribute__((ext_vector_type(8)))  int si8;
typedef __attribute__((ext_vector_type(4)))  int si4;

#define DEV static __device__ __forceinline__

DEV v16bf frag_from(v4u lo, v4u hi) {
  union { v4u u[2]; v16bf v; } x;
  x.u[0] = lo; x.u[1] = hi;
  return x.v;
}

// A-matrix fragment (16x32 bf16, M rows m0..m0+15, K cols k0..k0+31).
// ISA layout: lanes 0-15 = rows, VGPR0-3 hold K=(half*8)..+7, VGPR4-7 hold K=16+(half*8)..+7
DEV v16bf load_frag_a(const __bf16* base, int ld, int m0, int k0, int lane) {
  const __bf16* p = base + (size_t)(m0 + (lane & 15)) * ld + k0 + ((lane >> 4) << 3);
  return frag_from(*(const v4u*)p, *(const v4u*)(p + 16));
}

// B-matrix fragment (32x16 bf16). B = W^T where W is [N,K] row-major, so B column n = W row n.
// ISA layout: lane%16 = column (N), VGPR0-3 hold K=(half*8)..+7, VGPR4-7 hold K=16+(half*8)..+7
DEV v16bf load_frag_b(const __bf16* w, int ld, int n0, int k0, int lane) {
  const __bf16* p = w + (size_t)(n0 + (lane & 15)) * ld + k0 + ((lane >> 4) << 3);
  return frag_from(*(const v4u*)p, *(const v4u*)(p + 16));
}

DEV v8f wmma_bf16(v16bf a, v16bf b, v8f c) {
  return __builtin_amdgcn_wmma_f32_16x16x32_bf16(false, a, false, b, (short)0, c, false, false);
}

__global__ void convert_f32_bf16(const float* __restrict__ in, __bf16* __restrict__ out, size_t n) {
  size_t i = (size_t)blockIdx.x * blockDim.x + threadIdx.x;
  if (i < n) out[i] = (__bf16)in[i];
}

// Out[M,N] = A[M,K] * W[N,K]^T + bias[N].  Block = 256 threads = 8 waves.
// Wave tile 64x64 (4x4 WMMA tiles); waves arranged 2(M) x 4(N) -> WG tile 128x256.
template <typename OutT>
__global__ __launch_bounds__(256) void gemm_bf16_kernel(
    const __bf16* __restrict__ A, const __bf16* __restrict__ W,
    const float* __restrict__ bias, OutT* __restrict__ Out,
    int M, int N, int K) {
  int lane = threadIdx.x & 31;
  int wave = threadIdx.x >> 5;
  int wm = wave & 1, wn = wave >> 1;
  int m0 = blockIdx.y * 128 + wm * 64;
  int n0 = blockIdx.x * 256 + wn * 64;

  v8f acc[4][4] = {};
  for (int k0 = 0; k0 < K; k0 += 32) {
    v16bf af[4], bf[4];
#pragma unroll
    for (int i = 0; i < 4; i++) af[i] = load_frag_a(A, K, m0 + 16 * i, k0, lane);
#pragma unroll
    for (int j = 0; j < 4; j++) bf[j] = load_frag_b(W, K, n0 + 16 * j, k0, lane);
#pragma unroll
    for (int i = 0; i < 4; i++)
#pragma unroll
      for (int j = 0; j < 4; j++)
        acc[i][j] = wmma_bf16(af[i], bf[j], acc[i][j]);
  }

  int half = lane >> 4, cm = lane & 15;
#pragma unroll
  for (int j = 0; j < 4; j++) {
    float bj = bias[n0 + 16 * j + cm];
#pragma unroll
    for (int i = 0; i < 4; i++) {
#pragma unroll
      for (int e = 0; e < 8; e++) {
        size_t row = (size_t)(m0 + 16 * i + e + 8 * half);
        size_t col = (size_t)(n0 + 16 * j + cm);
        Out[row * (size_t)N + col] = (OutT)(acc[i][j][e] + bj);
      }
    }
  }
}

// Flash attention over qkv buffer [B*S, 6144] bf16 (Q at col h*128, K at 2048+h*128, V at 4096+h*128).
// Grid: (S/128, nh, B). Block 256 = 8 waves; wave handles 16 query rows.
// K tile is DMA'd into LDS by the Tensor Data Mover (one descriptor per key block, wave 0),
// then all 8 waves read B-fragments from LDS (row-major K tile == contiguous B columns).
// The K-tile buffer is reused as the P-staging buffer after the score WMMAs (barrier-separated).
__global__ __launch_bounds__(256) void attn_kernel(
    const __bf16* __restrict__ qkv, __bf16* __restrict__ obf) {
  const int S = 2048, LD = 6144, NH_HD = 2048;
  __shared__ __bf16 KtP[128 * 128];  // K tile (row-major [key][d]); later P staging. 32KB
  __shared__ __bf16 Vt[128 * 128];   // V tile transposed: Vt[d][key]. 32KB

  int lane = threadIdx.x & 31;
  int wave = threadIdx.x >> 5;
  int q0 = blockIdx.x * 128;
  int h = blockIdx.y;
  int bb = blockIdx.z;

  const __bf16* base = qkv + (size_t)bb * S * LD + h * 128;
  const __bf16* Qb = base;
  const __bf16* Kb = base + 2048;
  const __bf16* Vb = base + 4096;
  int qw = q0 + wave * 16;
  int half = lane >> 4, cm = lane & 15;
  const float scale = 0.08838834764831845f;  // 1/sqrt(128)

  v16bf qf[4];
#pragma unroll
  for (int t = 0; t < 4; t++) qf[t] = load_frag_a(Qb, LD, qw, t * 32, lane);

  v8f o[8] = {};
  float mi[8], li[8];
#pragma unroll
  for (int e = 0; e < 8; e++) { mi[e] = -INFINITY; li[e] = 0.0f; }

  int kbmax = q0 / 128;
  for (int kb = 0; kb <= kbmax; kb++) {
#if __has_builtin(__builtin_amdgcn_tensor_load_to_lds)
    // --- TDM: DMA K tile [128 keys x 128 d] (bf16, row stride 6144 elems) into LDS ---
    if (wave == 0) {
      unsigned long long gaddr =
          (unsigned long long)(uintptr_t)(Kb + (size_t)kb * 128 * LD);
      unsigned ldsa = (unsigned)(uintptr_t)(&KtP[0]);
      su4 g0;
      g0.x = 1u;                                   // count = 1 valid descriptor
      g0.y = ldsa;                                 // lds_addr [63:32]
      g0.z = (unsigned)(gaddr & 0xFFFFFFFFu);      // global_addr lo  [95:64]
      g0.w = (unsigned)((gaddr >> 32) & 0x1FFFFFFu) | 0x80000000u;  // addr hi + type=2
      si8 g1;
      g1[0] = 0x00010000;       // workgroup_mask=0, data_size=1 (2 bytes)
      g1[1] = (int)(128u << 16); // tensor_dim0 = 128 (lo16 at [63:48])
      g1[2] = (int)(128u << 16); // tensor_dim1 = 128 (lo16 at [95:80])
      g1[3] = (int)(128u << 16); // tile_dim0   = 128 ([127:112])
      g1[4] = 128;               // tile_dim1   = 128 ([143:128]), tile_dim2 = 0
      g1[5] = 6144;              // tensor_dim0_stride lo32 (elements)
      g1[6] = 0;                 // stride0 hi / tensor_dim1_stride lo
      g1[7] = 0;
      si4 z4 = {0, 0, 0, 0};
      si8 z8 = {0, 0, 0, 0, 0, 0, 0, 0};
      // 6-arg form (clang-23 / therock-10.0 headers): (g0, g1, g2, g3, g4, cpol)
      __builtin_amdgcn_tensor_load_to_lds(g0, g1, z4, z4, z8, 0);
    }
#else
    // Fallback: cooperative global->LDS copy of K tile (row-major, 16B chunks)
    for (int c = threadIdx.x; c < 128 * 16; c += 256) {
      int key = c >> 4;
      int dc = (c & 15) << 3;
      *(v4u*)(&KtP[key * 128 + dc]) =
          *(const v4u*)(Kb + (size_t)(kb * 128 + key) * LD + dc);
    }
#endif

    // --- Cooperatively stage V tile transposed into LDS: Vt[d][key] (overlaps the DMA) ---
    for (int c = threadIdx.x; c < 128 * 16; c += 256) {
      int key = c >> 4;
      int dc = (c & 15) << 3;
      const __bf16* vp = Vb + (size_t)(kb * 128 + key) * LD + dc;
      union { v4u u; __bf16 e[8]; } u;
      u.u = *(const v4u*)vp;
#pragma unroll
      for (int x = 0; x < 8; x++) Vt[(dc + x) * 128 + key] = u.e[x];
    }

#if __has_builtin(__builtin_amdgcn_tensor_load_to_lds)
    if (wave == 0) {
#if __has_builtin(__builtin_amdgcn_s_wait_tensorcnt)
      __builtin_amdgcn_s_wait_tensorcnt(0);
#else
      asm volatile("s_wait_tensorcnt 0x0" ::: "memory");
#endif
    }
#endif
    __syncthreads();

    // --- Scores: S = Q * K^T, K B-fragments from LDS tile (ld = 128) ---
    v8f s[8];
#pragma unroll
    for (int j = 0; j < 8; j++) {
      v8f acc = {};
#pragma unroll
      for (int t = 0; t < 4; t++) {
        v16bf kf = load_frag_b(KtP, 128, j * 16, t * 32, lane);
        acc = wmma_bf16(qf[t], kf, acc);
      }
      s[j] = acc;
    }

    bool diag = (kb == kbmax);
    float nm[8];
#pragma unroll
    for (int e = 0; e < 8; e++) nm[e] = mi[e];
#pragma unroll
    for (int j = 0; j < 8; j++) {
#pragma unroll
      for (int e = 0; e < 8; e++) {
        float v = s[j][e] * scale;
        if (diag) {
          int row = qw + e + 8 * half;
          int col = kb * 128 + j * 16 + cm;
          if (col > row) v = -INFINITY;
        }
        s[j][e] = v;
        nm[e] = fmaxf(nm[e], v);
      }
    }
    // Row max: reduce across the 16 lanes holding this row's columns
#pragma unroll
    for (int e = 0; e < 8; e++) {
      for (int off = 8; off >= 1; off >>= 1)
        nm[e] = fmaxf(nm[e], __shfl_xor(nm[e], off, 32));
    }

    __syncthreads();  // all waves done reading K tile; its LDS becomes P staging

    __bf16* Pw = &KtP[wave * 2048];  // 16x128 bf16 per-wave P slice
    float ls[8];
#pragma unroll
    for (int e = 0; e < 8; e++) ls[e] = 0.0f;
#pragma unroll
    for (int j = 0; j < 8; j++) {
#pragma unroll
      for (int e = 0; e < 8; e++) {
        float p = __expf(s[j][e] - nm[e]);
        ls[e] += p;
        Pw[(e + 8 * half) * 128 + j * 16 + cm] = (__bf16)p;
      }
    }
#pragma unroll
    for (int e = 0; e < 8; e++) {
      for (int off = 8; off >= 1; off >>= 1)
        ls[e] += __shfl_xor(ls[e], off, 32);
      float alpha = __expf(mi[e] - nm[e]);
      li[e] = li[e] * alpha + ls[e];
      mi[e] = nm[e];
#pragma unroll
      for (int j = 0; j < 8; j++) o[j][e] *= alpha;
    }
    __syncthreads();  // P visible (cross-lane) before A-fragment reads

    // --- O += P * V  (A = P from LDS, B = V^T columns contiguous in Vt) ---
    v16bf pf[4];
#pragma unroll
    for (int t = 0; t < 4; t++) pf[t] = load_frag_a(Pw, 128, 0, t * 32, lane);
#pragma unroll
    for (int j = 0; j < 8; j++) {
#pragma unroll
      for (int t = 0; t < 4; t++) {
        v16bf vf = load_frag_b(Vt, 128, j * 16, t * 32, lane);
        o[j] = wmma_bf16(pf[t], vf, o[j]);
      }
    }
    __syncthreads();  // done reading Vt/P before next tile overwrites them
  }

  // Normalize and store O as bf16: [B*S, 2048], col = h*128 + d
#pragma unroll
  for (int j = 0; j < 8; j++) {
#pragma unroll
    for (int e = 0; e < 8; e++) {
      float v = o[j][e] / li[e];
      size_t row = (size_t)bb * S + qw + e + 8 * half;
      int col = h * 128 + j * 16 + cm;
      obf[row * NH_HD + col] = (__bf16)v;
    }
  }
}

extern "C" void kernel_launch(void* const* d_in, const int* in_sizes, int n_in,
                              void* d_out, int out_size, void* d_ws, size_t ws_size,
                              hipStream_t stream) {
  (void)in_sizes; (void)n_in; (void)out_size; (void)ws_size;
  const float* x    = (const float*)d_in[0];
  const float* Wqkv = (const float*)d_in[1];
  const float* bqkv = (const float*)d_in[2];
  const float* Wout = (const float*)d_in[3];
  const float* bout = (const float*)d_in[4];
  float* out = (float*)d_out;

  const int B = 2, S = 2048, H = 2048;
  const int M = B * S;        // 4096
  const int N1 = 3 * H;       // 6144

  char* ws = (char*)d_ws;
  __bf16* x_bf    = (__bf16*)ws; ws += (size_t)M * H * 2;
  __bf16* wqkv_bf = (__bf16*)ws; ws += (size_t)N1 * H * 2;
  __bf16* wout_bf = (__bf16*)ws; ws += (size_t)H * H * 2;
  __bf16* qkv_bf  = (__bf16*)ws; ws += (size_t)M * N1 * 2;
  __bf16* o_bf    = (__bf16*)ws;

  {
    size_t n = (size_t)M * H;
    convert_f32_bf16<<<(unsigned)((n + 255) / 256), 256, 0, stream>>>(x, x_bf, n);
  }
  {
    size_t n = (size_t)N1 * H;
    convert_f32_bf16<<<(unsigned)((n + 255) / 256), 256, 0, stream>>>(Wqkv, wqkv_bf, n);
  }
  {
    size_t n = (size_t)H * H;
    convert_f32_bf16<<<(unsigned)((n + 255) / 256), 256, 0, stream>>>(Wout, wout_bf, n);
  }

  // QKV projection: [4096,6144] bf16
  gemm_bf16_kernel<__bf16><<<dim3(N1 / 256, M / 128), 256, 0, stream>>>(
      x_bf, wqkv_bf, bqkv, qkv_bf, M, N1, H);

  // Flash attention -> O bf16 [4096,2048]
  attn_kernel<<<dim3(S / 128, 16, B), 256, 0, stream>>>(qkv_bf, o_bf);

  // Output projection -> fp32 result
  gemm_bf16_kernel<float><<<dim3(H / 256, M / 128), 256, 0, stream>>>(
      o_bf, wout_bf, bout, out, M, H, H);
}
